// Procedure_69277822484745
// MI455X (gfx1250) — matrix operand order repaired and checked by reading, not compiled, not run on hardware
//
#include <hip/hip_runtime.h>

// ---------------------------------------------------------------------------
// Problem constants (from reference)
// ---------------------------------------------------------------------------
#define BB     2048
#define NN     200
#define TT     256       // type vocab / segment width
#define HH     256       // shop hidden
#define SS     16        // shop out
#define THE    64        // type emb dim
#define INF_   99        // shop MLP input features
#define KP     128       // padded K for GEMM1
#define ROWS   64        // rows per block
#define XSTR   136       // LDS row stride (halfs) for X  (272B, 16B aligned, bank-staggered)
#define HSTR   264       // LDS row stride (halfs) for H  (528B, 16B aligned, bank-staggered)

typedef __attribute__((ext_vector_type(16))) _Float16 v16h;
typedef __attribute__((ext_vector_type(8)))  _Float16 v8h;
typedef __attribute__((ext_vector_type(8)))  float    v8f;

// ---------------------------------------------------------------------------
// Input index map (jax pytree flatten: dict keys sorted)
// 0..10 raw inputs; params: cnn, combine, pair, rnn
// ---------------------------------------------------------------------------
enum {
  I_TYPES=0, I_REGIONS, I_RATINGS, I_PRICES, I_LEN,
  I_NTYPE, I_NDIST, I_NRAT, I_NCOM, I_NPRICE, I_NGRP,
  // cnn: conv(cb1,cb2,k1,k2), dist_emb, price_emb, score(W1,W2,b1,b2), shop(W1,W2,b1,b2), type_emb
  C_CB1=11, C_CB2, C_K1, C_K2, C_DIST, C_PRICE,
  C_SC_W1, C_SC_W2, C_SC_B1, C_SC_B2,
  C_SH_W1, C_SH_W2, C_SH_B1, C_SH_B2, C_TYPE,
  // combine
  M_W1=26, M_W2, M_B1, M_B2,
  // pair
  P_DIST=30, P_PRICE, P_SC_W1, P_SC_W2, P_SC_B1, P_SC_B2,
  P_SH_W1, P_SH_W2, P_SH_B1, P_SH_B2, P_TYPE,
  // rnn: dist_emb, lstm[3]{Wh,Wi,bh,bi}, price_emb, score, shop, type_emb
  R_DIST=41,
  R_L0_WH, R_L0_WI, R_L0_BH, R_L0_BI,
  R_L1_WH, R_L1_WI, R_L1_BH, R_L1_BI,
  R_L2_WH, R_L2_WI, R_L2_BH, R_L2_BI,
  R_PRICE=54, R_SC_W1, R_SC_W2, R_SC_B1, R_SC_B2,
  R_SH_W1, R_SH_W2, R_SH_B1, R_SH_B2, R_TYPE
};

// ---------------------------------------------------------------------------
// Workspace layout (bytes)
// ---------------------------------------------------------------------------
constexpr size_t PACK_W1  = (size_t)KP * HH * 2;    // 65536
constexpr size_t PACK_W2  = (size_t)HH * SS * 2;    // 8192
constexpr size_t PACK_BR  = PACK_W1 + PACK_W2;      // 73728 per branch (pair,cnn,rnn)
constexpr size_t OFF_SEG  = 3 * PACK_BR;
constexpr size_t OFF_CTX  = OFF_SEG  + (size_t)BB*TT*4;
constexpr size_t OFF_SSUM = OFF_CTX  + (size_t)BB*SS*4;
constexpr size_t OFF_CNT  = OFF_SSUM + (size_t)BB*4;
constexpr size_t OFF_H00  = OFF_CNT  + (size_t)BB*4;
constexpr size_t OFF_SEQM = OFF_H00  + 256;
constexpr size_t OFF_SEQF = OFF_SEQM + (size_t)BB*4;
constexpr size_t OFF_HL   = OFF_SEQF + (size_t)BB*4;
constexpr size_t OFF_PRES = OFF_HL   + (size_t)BB*SS*4;
constexpr size_t OFF_CRES = OFF_PRES + (size_t)BB*4;
constexpr size_t OFF_RRES = OFF_CRES + (size_t)BB*4;
constexpr size_t OFF_CTX2 = OFF_RRES + (size_t)BB*4;
constexpr size_t WS_NEED  = OFF_CTX2 + (size_t)BB*SS*4;

// ---------------------------------------------------------------------------
// Zero scratch (atomic accumulators)
// ---------------------------------------------------------------------------
__global__ void zero_kernel(unsigned int* p, int nwords) {
  int i = blockIdx.x * blockDim.x + threadIdx.x;
  for (; i < nwords; i += gridDim.x * blockDim.x) p[i] = 0u;
}

// ---------------------------------------------------------------------------
// Pack a KxN f32 weight into WMMA B-fragment order (f16), zero-padding K.
// Fragment (kk, j): lane l holds K = kk*32 + (l<16?0:8) + (t&7) + (t<8?0:16),
// n = j*16+(l&15), t = 0..15.
// ---------------------------------------------------------------------------
__global__ void pack_b_kernel(const float* __restrict__ W, _Float16* __restrict__ out,
                              int K32, int NT, int Ksrc, int Ncols) {
  int total = K32 * NT * 512;
  int i = blockIdx.x * blockDim.x + threadIdx.x;
  for (; i < total; i += gridDim.x * blockDim.x) {
    int kk  = i / (NT * 512);
    int rem = i % (NT * 512);
    int j   = rem / 512;
    int q   = rem % 512;
    int l   = q >> 4;
    int t   = q & 15;
    int n   = j * 16 + (l & 15);
    int k   = kk * 32 + ((l < 16) ? 0 : 8) + (t & 7) + ((t & 8) ? 16 : 0);
    float v = (k < Ksrc) ? W[k * Ncols + n] : 0.0f;
    out[i] = (_Float16)v;
  }
}

// ---------------------------------------------------------------------------
// Main fused shop-MLP GEMM (WMMA) + branch-specific reduction epilogue.
// mode: 0=pair (masked ctx sum), 1=cnn (segment sum), 2=rnn (dist==0 sum)
// ---------------------------------------------------------------------------
__global__ __launch_bounds__(128)
void shop_gemm_kernel(int mode,
    const int*   __restrict__ ntype,  const int*   __restrict__ ndist,
    const float* __restrict__ nrat,   const float* __restrict__ ncom,
    const int*   __restrict__ nprice, const float* __restrict__ ngrp,
    const int*   __restrict__ len,
    const float* __restrict__ type_emb, const float* __restrict__ dist_emb,
    const float* __restrict__ price_emb,
    const _Float16* __restrict__ W1p, const float* __restrict__ b1,
    const _Float16* __restrict__ W2p, const float* __restrict__ b2,
    float* __restrict__ ctx, float* __restrict__ seg,
    float* __restrict__ ssum, int* __restrict__ cnt, float* __restrict__ h00)
{
  __shared__ _Float16 sX[ROWS * XSTR];
  __shared__ _Float16 sH[ROWS * HSTR];

  const int tid  = threadIdx.x;
  const int lane = tid & 31;
  const int wave = tid >> 5;
  const int rowBase = blockIdx.x * ROWS;

  // ---- gather embeddings into padded f16 X tile in LDS -------------------
  for (int idx = tid; idx < ROWS * KP; idx += 128) {
    int r = idx >> 7;          // KP == 128
    int c = idx & 127;
    int g = rowBase + r;
    float v = 0.0f;
    if (c < 64)       { int t = ntype[g];  v = type_emb[t * 64 + c]; }
    else if (c < 80)  { int d = ndist[g];  v = dist_emb[d * 16 + (c - 64)]; }
    else if (c == 80) { v = nrat[g]; }
    else if (c == 81) { v = ncom[g]; }
    else if (c < 98)  { int p = nprice[g]; v = price_emb[p * 16 + (c - 82)]; }
    else if (c == 98) { v = ngrp[g]; }
    sX[r * XSTR + c] = (_Float16)v;
  }
  __syncthreads();

  const int half16 = (lane < 16) ? 0 : 8;  // A-frag K sub-offset / D row offset
  const int mrow   = lane & 15;
  const int wbase  = wave * 16;

  // ---- GEMM1: X(16x128) @ W1(128x256) ------------------------------------
  v8f acc[16];
  #pragma unroll
  for (int j = 0; j < 16; ++j) acc[j] = (v8f){0,0,0,0,0,0,0,0};

  #pragma unroll
  for (int kk = 0; kk < 4; ++kk) {
    const _Float16* ap = &sX[(wbase + mrow) * XSTR + kk * 32 + half16];
    v8h lo = *(const v8h*)ap;
    v8h hi = *(const v8h*)(ap + 16);
    v16h afrag;
    #pragma unroll
    for (int i = 0; i < 8; ++i) { afrag[i] = lo[i]; afrag[8 + i] = hi[i]; }
    #pragma unroll
    for (int j = 0; j < 16; ++j) {
      v16h bfrag = *(const v16h*)(W1p + (((kk * 16 + j) * 32 + lane) << 4));
      acc[j] = __builtin_amdgcn_wmma_f32_16x16x32_f16(
                   false, afrag, false, bfrag, (short)0, acc[j], false, false);
    }
  }

  // ---- bias + relu, stash h rows (wave-private) as f16 in LDS ------------
  #pragma unroll
  for (int j = 0; j < 16; ++j) {
    int col = j * 16 + mrow;
    float bb = b1[col];
    #pragma unroll
    for (int r = 0; r < 8; ++r) {
      float v = acc[j][r] + bb;
      v = v > 0.0f ? v : 0.0f;
      sH[(wbase + r + half16) * HSTR + col] = (_Float16)v;
    }
  }
  // no barrier: each wave consumes only the rows it produced (DScnt ordering)

  // ---- GEMM2: h(16x256) @ W2(256x16) -------------------------------------
  v8f d = (v8f){0,0,0,0,0,0,0,0};
  #pragma unroll
  for (int kk = 0; kk < 8; ++kk) {
    const _Float16* ap = &sH[(wbase + mrow) * HSTR + kk * 32 + half16];
    v8h lo = *(const v8h*)ap;
    v8h hi = *(const v8h*)(ap + 16);
    v16h afrag;
    #pragma unroll
    for (int i = 0; i < 8; ++i) { afrag[i] = lo[i]; afrag[8 + i] = hi[i]; }
    v16h bfrag = *(const v16h*)(W2p + ((kk * 32 + lane) << 4));
    d = __builtin_amdgcn_wmma_f32_16x16x32_f16(
            false, afrag, false, bfrag, (short)0, d, false, false);
  }

  // ---- branch epilogues ---------------------------------------------------
  const int col = mrow;
  const float bias2 = b2[col];

  if (mode == 0) {                      // pair: masked sum into ctx[b][col]
    #pragma unroll
    for (int r = 0; r < 8; ++r) {
      int g = rowBase + wbase + r + half16;
      int b = g / NN, n = g % NN;
      float v = d[r] + bias2;
      if (n < len[b]) atomicAdd(&ctx[b * SS + col], v);
    }
  } else if (mode == 1) {               // cnn: segment sum of row sums
    #pragma unroll
    for (int r = 0; r < 8; ++r) {
      int g = rowBase + wbase + r + half16;
      float v = d[r] + bias2;
      v += __shfl_xor(v, 1); v += __shfl_xor(v, 2);
      v += __shfl_xor(v, 4); v += __shfl_xor(v, 8);
      if (col == 0) {
        int b = g / NN;
        atomicAdd(&seg[b * TT + ntype[g]], v);
      }
    }
  } else {                              // rnn: dist==0 masked sums + row-0 capture
    #pragma unroll
    for (int r = 0; r < 8; ++r) {
      int g = rowBase + wbase + r + half16;
      float v = d[r] + bias2;
      if (g == 0) h00[col] = v;         // hid_r[0,0,:] (lanes 0-15 only)
      float s = v;
      s += __shfl_xor(s, 1); s += __shfl_xor(s, 2);
      s += __shfl_xor(s, 4); s += __shfl_xor(s, 8);
      if (col == 0 && ndist[g] == 0) {
        int b = g / NN;
        atomicAdd(&ssum[b], s);
        atomicAdd(&cnt[b], 1);
      }
    }
  }
}

// ---------------------------------------------------------------------------
// Score MLP: res = relu(concat(ctx16, te64) @ W1(80x64) + b1) @ W2(64) + b2
// mode 0: ctx = ctxbuf/max(len,1); mode 1/2: ctx = ctxbuf
// ---------------------------------------------------------------------------
__global__ __launch_bounds__(64)
void score_kernel(int mode, const float* __restrict__ ctxbuf,
                  const int* __restrict__ len, const int* __restrict__ types,
                  const float* __restrict__ type_emb,
                  const float* __restrict__ W1, const float* __restrict__ b1,
                  const float* __restrict__ W2, const float* __restrict__ b2,
                  float* __restrict__ res) {
  __shared__ float xs[80];
  __shared__ float red[64];
  int b = blockIdx.x, j = threadIdx.x;
  if (j < 16) {
    float c = ctxbuf[b * SS + j];
    if (mode == 0) {
      int L = len[b]; if (L < 1) L = 1;
      c /= (float)L;
    }
    xs[j] = c;
  }
  xs[16 + j] = type_emb[types[b] * THE + j];
  __syncthreads();
  float h = b1[j];
  #pragma unroll 4
  for (int k = 0; k < 80; ++k) h += xs[k] * W1[k * 64 + j];
  h = h > 0.0f ? h : 0.0f;
  red[j] = h * W2[j];
  __syncthreads();
  for (int off = 32; off > 0; off >>= 1) {
    if (j < off) red[j] += red[j + off];
    __syncthreads();
  }
  if (j == 0) res[b] = red[0] + b2[0];
}

// ---------------------------------------------------------------------------
// CNN: two 3x3 SAME convs on tmat[b,t,s]=seg[b,t]; only row types[b] needed.
// ---------------------------------------------------------------------------
__global__ void conv_ctx_kernel(const float* __restrict__ seg,
                                const int* __restrict__ types,
                                const float* __restrict__ k1, const float* __restrict__ cb1,
                                const float* __restrict__ k2, const float* __restrict__ cb2,
                                float* __restrict__ ctx2) {
  int id = blockIdx.x * blockDim.x + threadIdx.x;
  if (id >= BB * SS) return;
  int b = id / SS, s = id % SS;
  int t0 = types[b];
  float a2 = cb2[0];
  #pragma unroll
  for (int dt = -1; dt <= 1; ++dt) {
    int tt = t0 + dt;
    #pragma unroll
    for (int ds = -1; ds <= 1; ++ds) {
      int sp = s + ds;
      float x1 = 0.0f;
      if (tt >= 0 && tt < TT && sp >= 0 && sp < SS) {
        float a1 = cb1[0];
        #pragma unroll
        for (int d2 = -1; d2 <= 1; ++d2) {
          int t2 = tt + d2;
          if (t2 < 0 || t2 >= TT) continue;
          float sv = seg[b * TT + t2];
          #pragma unroll
          for (int e2 = -1; e2 <= 1; ++e2) {
            int s2 = sp + e2;
            if (s2 < 0 || s2 >= SS) continue;
            a1 += sv * k1[(d2 + 1) * 3 + (e2 + 1)];
          }
        }
        x1 = a1 > 0.0f ? a1 : 0.0f;
      }
      a2 += x1 * k2[(dt + 1) * 3 + (ds + 1)];
    }
  }
  ctx2[id] = a2 > 0.0f ? a2 : 0.0f;
}

// ---------------------------------------------------------------------------
// RNN: sequential "last valid mean" scan over B (trivial length)
// ---------------------------------------------------------------------------
__global__ void rnn_scan_kernel(const float* __restrict__ ssum, const int* __restrict__ cnt,
                                float* __restrict__ seqm, int* __restrict__ seqf) {
  if (blockIdx.x != 0 || threadIdx.x != 0) return;
  float last = 0.0f; int found = 0;
  for (int b = 0; b < BB; ++b) {
    int c = cnt[b];
    if (c > 0) { last = ssum[b] / ((float)c * (float)SS); found = 1; }
    seqm[b] = last;
    seqf[b] = found;
  }
}

__device__ __forceinline__ float sigm(float x) { return 1.0f / (1.0f + __expf(-x)); }

// RNN gate stack (reference never uses Wh): h' = sig(o)*tanh(sig(i)*tanh(g))
__global__ void rnn_lstm_kernel(const float* __restrict__ seqm, const int* __restrict__ seqf,
                                const float* __restrict__ h00,
                                const float* __restrict__ Wi0, const float* __restrict__ bi0, const float* __restrict__ bh0,
                                const float* __restrict__ Wi1, const float* __restrict__ bi1, const float* __restrict__ bh1,
                                const float* __restrict__ Wi2, const float* __restrict__ bi2, const float* __restrict__ bh2,
                                float* __restrict__ hout) {
  int b = blockIdx.x * blockDim.x + threadIdx.x;
  if (b >= BB) return;
  float h[SS];
  if (seqf[b]) { float m = seqm[b]; for (int j = 0; j < SS; ++j) h[j] = m; }
  else         { for (int j = 0; j < SS; ++j) h[j] = h00[j]; }

  const float* Wi[3] = {Wi0, Wi1, Wi2};
  const float* bi[3] = {bi0, bi1, bi2};
  const float* bh[3] = {bh0, bh1, bh2};
  for (int L = 0; L < 3; ++L) {
    float g[64];
    for (int j = 0; j < 64; ++j) {
      float a = bi[L][j] + bh[L][j];
      #pragma unroll
      for (int k = 0; k < SS; ++k) a += h[k] * Wi[L][k * 64 + j];
      g[j] = a;
    }
    for (int j = 0; j < SS; ++j) {
      float c = sigm(g[j]) * tanhf(g[32 + j]);
      h[j] = sigm(g[48 + j]) * tanhf(c);
    }
  }
  for (int j = 0; j < SS; ++j) hout[b * SS + j] = h[j];
}

// ---------------------------------------------------------------------------
// Combine: out = relu([p,c,r] @ W1(3x16) + b1) @ W2(16) + b2
// ---------------------------------------------------------------------------
__global__ void combine_kernel(const float* __restrict__ pres, const float* __restrict__ cres,
                               const float* __restrict__ rres,
                               const float* __restrict__ W1, const float* __restrict__ b1,
                               const float* __restrict__ W2, const float* __restrict__ b2,
                               float* __restrict__ out) {
  int b = blockIdx.x * blockDim.x + threadIdx.x;
  if (b >= BB) return;
  float z0 = pres[b], z1 = cres[b], z2 = rres[b];
  float s = 0.0f;
  #pragma unroll
  for (int j = 0; j < 16; ++j) {
    float a = b1[j] + z0 * W1[j] + z1 * W1[16 + j] + z2 * W1[32 + j];
    a = a > 0.0f ? a : 0.0f;
    s += a * W2[j];
  }
  out[b] = s + b2[0];
}

// ---------------------------------------------------------------------------
// Launch
// ---------------------------------------------------------------------------
extern "C" void kernel_launch(void* const* d_in, const int* in_sizes, int n_in,
                              void* d_out, int out_size, void* d_ws, size_t ws_size,
                              hipStream_t stream) {
  if (ws_size < WS_NEED) return;
  char* ws = (char*)d_ws;

  const int*   types  = (const int*)  d_in[I_TYPES];
  const int*   len    = (const int*)  d_in[I_LEN];
  const int*   ntype  = (const int*)  d_in[I_NTYPE];
  const int*   ndist  = (const int*)  d_in[I_NDIST];
  const float* nrat   = (const float*)d_in[I_NRAT];
  const float* ncom   = (const float*)d_in[I_NCOM];
  const int*   nprice = (const int*)  d_in[I_NPRICE];
  const float* ngrp   = (const float*)d_in[I_NGRP];

  _Float16* W1p[3] = { (_Float16*)(ws + 0 * PACK_BR),
                       (_Float16*)(ws + 1 * PACK_BR),
                       (_Float16*)(ws + 2 * PACK_BR) };
  _Float16* W2p[3] = { (_Float16*)(ws + 0 * PACK_BR + PACK_W1),
                       (_Float16*)(ws + 1 * PACK_BR + PACK_W1),
                       (_Float16*)(ws + 2 * PACK_BR + PACK_W1) };

  float* seg   = (float*)(ws + OFF_SEG);
  float* ctx   = (float*)(ws + OFF_CTX);
  float* ssum  = (float*)(ws + OFF_SSUM);
  int*   cnt   = (int*)  (ws + OFF_CNT);
  float* h00   = (float*)(ws + OFF_H00);
  float* seqm  = (float*)(ws + OFF_SEQM);
  int*   seqf  = (int*)  (ws + OFF_SEQF);
  float* hl    = (float*)(ws + OFF_HL);
  float* pres  = (float*)(ws + OFF_PRES);
  float* cres  = (float*)(ws + OFF_CRES);
  float* rres  = (float*)(ws + OFF_RRES);
  float* ctx2  = (float*)(ws + OFF_CTX2);

  // 1) zero atomic accumulators
  {
    int nwords = (int)((OFF_H00 + 256 - OFF_SEG) / 4);
    zero_kernel<<<512, 256, 0, stream>>>((unsigned int*)(ws + OFF_SEG), nwords);
  }

  // 2) pack weights to f16 B-fragment order (branch order: pair, cnn, rnn)
  const float* shW1[3] = { (const float*)d_in[P_SH_W1], (const float*)d_in[C_SH_W1], (const float*)d_in[R_SH_W1] };
  const float* shW2[3] = { (const float*)d_in[P_SH_W2], (const float*)d_in[C_SH_W2], (const float*)d_in[R_SH_W2] };
  for (int br = 0; br < 3; ++br) {
    pack_b_kernel<<<(KP * HH + 255) / 256, 256, 0, stream>>>(shW1[br], W1p[br], KP / 32, HH / 16, INF_, HH);
    pack_b_kernel<<<(HH * SS + 255) / 256, 256, 0, stream>>>(shW2[br], W2p[br], HH / 32, SS / 16, HH, SS);
  }

  // 3) fused WMMA shop-MLP + reduction, per branch
  const float* te[3]  = { (const float*)d_in[P_TYPE],  (const float*)d_in[C_TYPE],  (const float*)d_in[R_TYPE] };
  const float* de[3]  = { (const float*)d_in[P_DIST],  (const float*)d_in[C_DIST],  (const float*)d_in[R_DIST] };
  const float* pe[3]  = { (const float*)d_in[P_PRICE], (const float*)d_in[C_PRICE], (const float*)d_in[R_PRICE] };
  const float* b1v[3] = { (const float*)d_in[P_SH_B1], (const float*)d_in[C_SH_B1], (const float*)d_in[R_SH_B1] };
  const float* b2v[3] = { (const float*)d_in[P_SH_B2], (const float*)d_in[C_SH_B2], (const float*)d_in[R_SH_B2] };

  int gemmBlocks = (BB * NN) / ROWS;   // 6400
  for (int br = 0; br < 3; ++br) {
    shop_gemm_kernel<<<gemmBlocks, 128, 0, stream>>>(
        br, ntype, ndist, nrat, ncom, nprice, ngrp, len,
        te[br], de[br], pe[br],
        W1p[br], b1v[br], W2p[br], b2v[br],
        ctx, seg, ssum, cnt, h00);
  }

  // 4) pair score
  score_kernel<<<BB, 64, 0, stream>>>(0, ctx, len, types, te[0],
      (const float*)d_in[P_SC_W1], (const float*)d_in[P_SC_B1],
      (const float*)d_in[P_SC_W2], (const float*)d_in[P_SC_B2], pres);

  // 5) cnn conv + score
  conv_ctx_kernel<<<(BB * SS + 127) / 128, 128, 0, stream>>>(
      seg, types,
      (const float*)d_in[C_K1], (const float*)d_in[C_CB1],
      (const float*)d_in[C_K2], (const float*)d_in[C_CB2], ctx2);
  score_kernel<<<BB, 64, 0, stream>>>(1, ctx2, len, types, te[1],
      (const float*)d_in[C_SC_W1], (const float*)d_in[C_SC_B1],
      (const float*)d_in[C_SC_W2], (const float*)d_in[C_SC_B2], cres);

  // 6) rnn scan + gate stack + score
  rnn_scan_kernel<<<1, 1, 0, stream>>>(ssum, cnt, seqm, seqf);
  rnn_lstm_kernel<<<(BB + 127) / 128, 128, 0, stream>>>(
      seqm, seqf, h00,
      (const float*)d_in[R_L0_WI], (const float*)d_in[R_L0_BI], (const float*)d_in[R_L0_BH],
      (const float*)d_in[R_L1_WI], (const float*)d_in[R_L1_BI], (const float*)d_in[R_L1_BH],
      (const float*)d_in[R_L2_WI], (const float*)d_in[R_L2_BI], (const float*)d_in[R_L2_BH],
      hl);
  score_kernel<<<BB, 64, 0, stream>>>(2, hl, len, types, te[2],
      (const float*)d_in[R_SC_W1], (const float*)d_in[R_SC_B1],
      (const float*)d_in[R_SC_W2], (const float*)d_in[R_SC_B2], rres);

  // 7) combine -> out
  combine_kernel<<<(BB + 127) / 128, 128, 0, stream>>>(
      pres, cres, rres,
      (const float*)d_in[M_W1], (const float*)d_in[M_B1],
      (const float*)d_in[M_W2], (const float*)d_in[M_B2],
      (float*)d_out);
}
